// Former_11433202942082
// MI455X (gfx1250) — matrix operand order repaired and checked by reading, not compiled, not run on hardware
//
#include <hip/hip_runtime.h>
#include <math.h>

// ---------------- problem constants ----------------
#define BN   16
#define CC   64
#define HH   128
#define WWD  128
#define HWL  16384            // 128*128
#define PLANE ((size_t)BN * CC * HWL)   // 16,777,216 floats = 64 MiB

typedef __attribute__((ext_vector_type(16))) _Float16 v16h;
typedef __attribute__((ext_vector_type(8)))  _Float16 v8h;
typedef __attribute__((ext_vector_type(8)))  float    v8f;

#if __has_builtin(__builtin_amdgcn_global_load_async_to_lds_b128) && \
    __has_builtin(__builtin_amdgcn_s_wait_asynccnt)
#define USE_ASYNC_LDS 1
// generic (gcc-style) vector to match the builtin's parameter type exactly
typedef int gv4i __attribute__((vector_size(16)));
typedef __attribute__((address_space(1))) gv4i* async_gptr_t;   // global (__device__)
typedef __attribute__((address_space(3))) gv4i* async_lptr_t;   // LDS (__shared__)
#endif

// ---------------- helpers ----------------
__device__ __forceinline__ float wave_reduce(float v) {
#pragma unroll
  for (int o = 16; o > 0; o >>= 1) v += __shfl_down(v, o, 32);
  return v;
}

__device__ __forceinline__ int imin(int a, int b) { return a < b ? a : b; }

// =====================================================================
// LayerNorm over C=64 (NCHW): one thread per (b, spatial) position.
// =====================================================================
__global__ void k_layernorm(const float* __restrict__ X, const float* __restrict__ g,
                            const float* __restrict__ bta, float* __restrict__ Y) {
  long i = (long)blockIdx.x * 256 + threadIdx.x;
  if (i >= (long)BN * HWL) return;
  int pos = (int)(i % HWL);
  int b   = (int)(i / HWL);
  const float* xp = X + (size_t)b * CC * HWL + pos;
  float v[CC];
  float mean = 0.f;
#pragma unroll
  for (int c = 0; c < CC; ++c) { v[c] = xp[(size_t)c * HWL]; mean += v[c]; }
  mean *= (1.f / CC);
  float var = 0.f;
#pragma unroll
  for (int c = 0; c < CC; ++c) { float d = v[c] - mean; var += d * d; }
  var *= (1.f / CC);
  float inv = rsqrtf(var + 1e-5f);
  float* yp = Y + (size_t)b * CC * HWL + pos;
#pragma unroll
  for (int c = 0; c < CC; ++c) yp[(size_t)c * HWL] = (v[c] - mean) * inv * g[c] + bta[c];
}

// =====================================================================
// 1x1 conv == GEMM  Y[b] = W(Cout x Cin) * X[b](Cin x HW) (+bias)(+act)
// WMMA f32_16x16x32_f16, K-step 64 (2 WMMA / iter). Block: 256 thr = 8
// waves, block tile M=16 x N=128 (one 16x16 WMMA tile per wave, A shared).
// Pipeline per K-step:
//   async global->LDS B128 (raw f32, landing zone sX)  [fallback: float4]
//   -> tile convert/transpose to f16 K-major sBt[n][k] (mask applied once)
//   -> fragments are CONTIGUOUS vector LDS loads (2x ds_load_b128 for B)
// act: 0=none 1=relu 2=sigmoid. Output supports channel-concat placement.
// =====================================================================
#define BT_STRIDE 80   // halves per sBt row (64 + 16 pad: kills bank stride)
#define SA_STRIDE 72   // halves per sA  row (64 + 8 pad)

__global__ __launch_bounds__(256)
void k_gemm1x1(const float* __restrict__ X, const float* __restrict__ W,
               const float* __restrict__ bias, float* __restrict__ Y,
               int Cin, int Cout, int out_ctot, int out_coff, int act) {
  __shared__ __align__(16) _Float16 sA[16 * SA_STRIDE];    // W tile (M x K) f16
  __shared__ __align__(32) _Float16 sBt[128 * BT_STRIDE];  // X tile transposed (N x K) f16
#ifdef USE_ASYNC_LDS
  __shared__ __align__(16) float sX[64 * 128];             // raw f32 landing zone
#endif
  const int tid  = threadIdx.x;
  const int lane = tid & 31;
  const int wv   = tid >> 5;
  const int n0   = blockIdx.x * 128;
  const int m0   = blockIdx.y * 16;
  const int b    = blockIdx.z;
  const float* Xb = X + (size_t)b * Cin * HWL;

  v8f acc = {0.f, 0.f, 0.f, 0.f, 0.f, 0.f, 0.f, 0.f};
  const int Kt = (Cin + 63) >> 6;

#pragma unroll 2
  for (int kt = 0; kt < Kt; ++kt) {
    const int k0 = kt << 6;

#ifdef USE_ASYNC_LDS
    // ---- async-stage X tile rows k0..k0+63 (64x128 f32, B128 per lane) ----
#pragma unroll
    for (int r = 0; r < 8; ++r) {
      int f4 = tid + r * 256;
      int kk = f4 >> 5;                    // row 0..63
      int c4 = f4 & 31;                    // float4 within row
      int krow = imin(k0 + kk, Cin - 1);   // clamp: always a valid address
      const float* src = Xb + (size_t)krow * HWL + n0 + c4 * 4;
      __builtin_amdgcn_global_load_async_to_lds_b128(
          (async_gptr_t)src, (async_lptr_t)&sX[kk * 128 + c4 * 4], 0, 0);
    }
#endif

    // ---- stage W tile (16x64 f16, zero-padded, branch-free) ----
#pragma unroll
    for (int r = 0; r < 4; ++r) {
      int e = tid + r * 256;               // 1024 elems
      int m = e >> 6, kk = e & 63;
      int mi = imin(m0 + m, Cout - 1);
      int ki = imin(k0 + kk, Cin - 1);
      float w = W[(size_t)mi * Cin + ki];
      w = (m0 + m < Cout && k0 + kk < Cin) ? w : 0.f;
      sA[m * SA_STRIDE + kk] = (_Float16)w;
    }
    // prefetch next K tile (global_prefetch_b8)
    if (k0 + 64 < Cin) __builtin_prefetch(&Xb[(size_t)(k0 + 64) * HWL + n0 + tid], 0, 1);

#ifdef USE_ASYNC_LDS
    __builtin_amdgcn_s_wait_asynccnt(0);
    __syncthreads();
    // ---- convert/transpose: sX f32 row-major -> sBt f16 K-major ----
#pragma unroll
    for (int r = 0; r < 8; ++r) {
      int f4 = tid + r * 256;
      int kk = f4 >> 5;
      int c4 = f4 & 31;
      const float4 v4 = *(const float4*)&sX[kk * 128 + c4 * 4];
      const bool ok = (k0 + kk) < Cin;
      int n = c4 * 4;
      sBt[(n + 0) * BT_STRIDE + kk] = (_Float16)(ok ? v4.x : 0.f);
      sBt[(n + 1) * BT_STRIDE + kk] = (_Float16)(ok ? v4.y : 0.f);
      sBt[(n + 2) * BT_STRIDE + kk] = (_Float16)(ok ? v4.z : 0.f);
      sBt[(n + 3) * BT_STRIDE + kk] = (_Float16)(ok ? v4.w : 0.f);
    }
#else
    // ---- fallback: global float4 -> convert -> sBt directly ----
#pragma unroll
    for (int r = 0; r < 8; ++r) {
      int f4 = tid + r * 256;
      int kk = f4 >> 5;
      int c4 = f4 & 31;
      int krow = imin(k0 + kk, Cin - 1);
      const float4 v4 = *(const float4*)(Xb + (size_t)krow * HWL + n0 + c4 * 4);
      const bool ok = (k0 + kk) < Cin;
      int n = c4 * 4;
      sBt[(n + 0) * BT_STRIDE + kk] = (_Float16)(ok ? v4.x : 0.f);
      sBt[(n + 1) * BT_STRIDE + kk] = (_Float16)(ok ? v4.y : 0.f);
      sBt[(n + 2) * BT_STRIDE + kk] = (_Float16)(ok ? v4.z : 0.f);
      sBt[(n + 3) * BT_STRIDE + kk] = (_Float16)(ok ? v4.w : 0.f);
    }
#endif
    __syncthreads();

    // ---- 2 WMMA per K-step; fragments are contiguous vector LDS loads ----
    const int arow = (lane & 15) * SA_STRIDE;
    const int nrow = (wv * 16 + (lane & 15)) * BT_STRIDE;
#pragma unroll
    for (int c = 0; c < 2; ++c) {
      // A: halves 0-7 -> K base..base+7 ; halves 8-15 -> K base+8..base+15
      const int abase = ((lane & 16) ? 16 : 0) + c * 32;
      const v8h alo = *(const v8h*)&sA[arow + abase];
      const v8h ahi = *(const v8h*)&sA[arow + abase + 8];
      v16h af;
#pragma unroll
      for (int j = 0; j < 8; ++j) { af[j] = alo[j]; af[j + 8] = ahi[j]; }
      // B: lane's 16 halves are 16 CONSECUTIVE K at fixed N
      const int bbase = ((lane & 16) ? 16 : 0) + c * 32;
      const v16h bf = *(const v16h*)&sBt[nrow + bbase];
      acc = __builtin_amdgcn_wmma_f32_16x16x32_f16(false, af, false, bf,
                                                   (short)0, acc, false, false);
    }
    __syncthreads();
  }

  // ---- epilogue: C/D layout lanes0-15 -> M=v, lanes16-31 -> M=v+8; N = lane&15
  const int ncol = n0 + wv * 16 + (lane & 15);
#pragma unroll
  for (int v = 0; v < 8; ++v) {
    int m = m0 + v + ((lane & 16) ? 8 : 0);
    if (m < Cout) {
      float val = acc[v];
      if (bias) val += bias[m];
      if (act == 1) val = fmaxf(val, 0.f);
      else if (act == 2) val = 1.f / (1.f + __expf(-val));
      Y[((size_t)b * out_ctot + out_coff + m) * HWL + ncol] = val;
    }
  }
}

// =====================================================================
// Grouped depthwise-ish conv (cpg input channels per group), k x k, pad.
// =====================================================================
__global__ void k_dwconv(const float* __restrict__ X, const float* __restrict__ Wd,
                         const float* __restrict__ bias, float* __restrict__ Y,
                         int Cin, int Cout, int cpg, int ks, int pad,
                         int out_ctot, int out_coff) {
  long i = (long)blockIdx.x * 256 + threadIdx.x;
  long total = (long)BN * Cout * HWL;
  if (i >= total) return;
  int pos = (int)(i % HWL);
  long r2 = i / HWL;
  int o = (int)(r2 % Cout);
  int b = (int)(r2 / Cout);
  int h = pos >> 7, wc = pos & 127;
  int G = Cin / cpg;
  int opg = Cout / G;
  int grp = o / opg;
  float acc = bias ? bias[o] : 0.f;
  for (int j = 0; j < cpg; ++j) {
    const float* xp = X + ((size_t)b * Cin + grp * cpg + j) * HWL;
    const float* wp = Wd + ((size_t)o * cpg + j) * ks * ks;
    for (int kh = 0; kh < ks; ++kh) {
      int hh = h + kh - pad;
      if (hh < 0 || hh >= HH) continue;
      for (int kw = 0; kw < ks; ++kw) {
        int wwp = wc + kw - pad;
        if (wwp < 0 || wwp >= WWD) continue;
        acc += xp[hh * WWD + wwp] * wp[kh * ks + kw];
      }
    }
  }
  Y[((size_t)b * out_ctot + out_coff + o) * HWL + pos] = acc;
}

// =====================================================================
// 128-pt Stockham radix-2 FFT in LDS. 4 lines/block, 64 thr/line.
// axis 0: along W (contiguous); axis 1: along H (stride 128).
// dir=+1 fwd, -1 inv (caller supplies 1/128 scale per inverse pass).
// mode 0: complex out; mode 1: |z|*scale into concat-placed real plane.
// =====================================================================
__global__ __launch_bounds__(256)
void k_fft128(const float* __restrict__ in_re, const float* __restrict__ in_im,
              float* __restrict__ out_re, float* __restrict__ out_im,
              int Cimg, int axis, int dir, float scale, int mode,
              int out_ctot, int out_coff) {
  __shared__ float2 buf[4][2][128];
  const int tid  = threadIdx.x;
  const int slot = tid >> 6;
  const int t    = tid & 63;
  const long line = (long)blockIdx.x * 4 + slot;
  const int img = (int)(line >> 7);
  const int l   = (int)(line & 127);
  const int bi  = img / Cimg, c = img % Cimg;
  const size_t ibase = (size_t)img * HWL;

#pragma unroll
  for (int i = 0; i < 2; ++i) {
    int p = t + i * 64;
    size_t off = (axis == 0) ? (size_t)l * 128 + p : (size_t)p * 128 + l;
    float re = in_re[ibase + off];
    float iv = in_im ? in_im[ibase + off] : 0.f;
    buf[slot][0][p] = make_float2(re, iv);
  }
  __syncthreads();

  int pb = 0;
  for (int st = 0; st < 7; ++st) {
    const int s = 1 << st;
    const int m = 64 >> st;
    const int n = 128 >> st;
    const int p = t / s;
    const int q = t - p * s;
    float ang = -6.28318530717958647f * (float)dir * (float)p / (float)n;
    float wi, wr;
    __sincosf(ang, &wi, &wr);
    float2 a  = buf[slot][pb][q + s * p];
    float2 bv = buf[slot][pb][q + s * (p + m)];
    float ur = a.x - bv.x, ui = a.y - bv.y;
    buf[slot][pb ^ 1][q + s * (2 * p)]     = make_float2(a.x + bv.x, a.y + bv.y);
    buf[slot][pb ^ 1][q + s * (2 * p + 1)] = make_float2(ur * wr - ui * wi, ur * wi + ui * wr);
    pb ^= 1;
    __syncthreads();
  }

  const size_t obase = ((size_t)bi * out_ctot + out_coff + c) * HWL;
#pragma unroll
  for (int i = 0; i < 2; ++i) {
    int p = t + i * 64;
    size_t off = (axis == 0) ? (size_t)l * 128 + p : (size_t)p * 128 + l;
    float2 z = buf[slot][pb][p];
    if (mode == 0) {
      out_re[obase + off] = z.x * scale;
      if (out_im) out_im[obase + off] = z.y * scale;
    } else {
      out_re[obase + off] = sqrtf(z.x * z.x + z.y * z.y) * scale;
    }
  }
}

// =====================================================================
// SSA channel attention: per (b, head), d = 8. One WG (256 thr) each.
// q,k l2norm over n, 8x8 gram via deterministic shfl+LDS tree reduce,
// softmax(temp * qk^T), out = attn @ v into concat slot of Y.
// =====================================================================
__global__ __launch_bounds__(256)
void k_ssa_attn(const float* __restrict__ qkv, const float* __restrict__ temp,
                float* __restrict__ Y, int out_ctot, int out_coff) {
  const int b  = blockIdx.x >> 3;
  const int hd = blockIdx.x & 7;
  const int tid = threadIdx.x, lane = tid & 31, wid = tid >> 5;
  const float* qp[8]; const float* kp[8]; const float* vp[8];
#pragma unroll
  for (int d = 0; d < 8; ++d) {
    qp[d] = qkv + ((size_t)b * 192 +   0 + hd * 8 + d) * HWL;
    kp[d] = qkv + ((size_t)b * 192 +  64 + hd * 8 + d) * HWL;
    vp[d] = qkv + ((size_t)b * 192 + 128 + hd * 8 + d) * HWL;
  }
  __shared__ float s_part[8][64];
  __shared__ float s_g[64], s_a[64], s_nq[8], s_nk[8];

  float accq[8], acck[8], g[64];
#pragma unroll
  for (int i = 0; i < 8; ++i) { accq[i] = 0.f; acck[i] = 0.f; }
#pragma unroll
  for (int i = 0; i < 64; ++i) g[i] = 0.f;

  for (int n = tid; n < HWL; n += 256) {
    float qv[8], kv[8];
#pragma unroll
    for (int d = 0; d < 8; ++d) { qv[d] = qp[d][n]; kv[d] = kp[d][n]; }
#pragma unroll
    for (int d = 0; d < 8; ++d) {
      accq[d] += qv[d] * qv[d];
      acck[d] += kv[d] * kv[d];
#pragma unroll
      for (int e = 0; e < 8; ++e) g[d * 8 + e] += qv[d] * kv[e];
    }
  }
  // gram reduce (deterministic: wave shfl tree then 8-way LDS sum)
#pragma unroll
  for (int i = 0; i < 64; ++i) {
    float r = wave_reduce(g[i]);
    if (lane == 0) s_part[wid][i] = r;
  }
  __syncthreads();
  if (tid < 64) { float s = 0.f; for (int w = 0; w < 8; ++w) s += s_part[w][tid]; s_g[tid] = s; }
  __syncthreads();
  // norm reduce
#pragma unroll
  for (int i = 0; i < 8; ++i) {
    float r = wave_reduce(accq[i]); if (lane == 0) s_part[wid][i] = r;
    r = wave_reduce(acck[i]);       if (lane == 0) s_part[wid][i + 8] = r;
  }
  __syncthreads();
  if (tid < 16) {
    float s = 0.f; for (int w = 0; w < 8; ++w) s += s_part[w][tid];
    s = fmaxf(sqrtf(s), 1e-12f);
    if (tid < 8) s_nq[tid] = s; else s_nk[tid - 8] = s;
  }
  __syncthreads();
  if (tid < 8) {
    const int d = tid; const float tmp = temp[hd];
    float row[8], mx = -1e30f;
#pragma unroll
    for (int e = 0; e < 8; ++e) { row[e] = s_g[d * 8 + e] / (s_nq[d] * s_nk[e]) * tmp; mx = fmaxf(mx, row[e]); }
    float sum = 0.f;
#pragma unroll
    for (int e = 0; e < 8; ++e) { row[e] = __expf(row[e] - mx); sum += row[e]; }
#pragma unroll
    for (int e = 0; e < 8; ++e) s_a[d * 8 + e] = row[e] / sum;
  }
  __syncthreads();
  for (int n = tid; n < HWL; n += 256) {
    float vv[8];
#pragma unroll
    for (int d = 0; d < 8; ++d) vv[d] = vp[d][n];
#pragma unroll
    for (int d = 0; d < 8; ++d) {
      float o = 0.f;
#pragma unroll
      for (int e = 0; e < 8; ++e) o += s_a[d * 8 + e] * vv[e];
      Y[((size_t)b * out_ctot + out_coff + hd * 8 + d) * HWL + n] = o;
    }
  }
}

// =====================================================================
// FSA frequency attention (complex, per (b, head), d=8): l2norm rows of v,
// complex gram, split softmax(real)+i*softmax(imag), out = attn @ v.
// =====================================================================
__global__ __launch_bounds__(256)
void k_fsa_attn(const float* __restrict__ re, const float* __restrict__ im,
                const float* __restrict__ temp,
                float* __restrict__ ore, float* __restrict__ oim) {
  const int b  = blockIdx.x >> 3;
  const int hd = blockIdx.x & 7;
  const int tid = threadIdx.x, lane = tid & 31, wid = tid >> 5;
  const float* rp[8]; const float* ip[8];
#pragma unroll
  for (int d = 0; d < 8; ++d) {
    rp[d] = re + ((size_t)b * CC + hd * 8 + d) * HWL;
    ip[d] = im + ((size_t)b * CC + hd * 8 + d) * HWL;
  }
  __shared__ float s_part[8][64];
  __shared__ float s_gr[64], s_gi[64], s_ar[64], s_ai[64], s_n[8];

  float gr[64], gi[64], nn[8];
#pragma unroll
  for (int i = 0; i < 64; ++i) { gr[i] = 0.f; gi[i] = 0.f; }
#pragma unroll
  for (int i = 0; i < 8; ++i) nn[i] = 0.f;

  for (int n = tid; n < HWL; n += 256) {
    float vr[8], vi[8];
#pragma unroll
    for (int d = 0; d < 8; ++d) { vr[d] = rp[d][n]; vi[d] = ip[d][n]; }
#pragma unroll
    for (int d = 0; d < 8; ++d) {
      nn[d] += vr[d] * vr[d] + vi[d] * vi[d];
#pragma unroll
      for (int e = 0; e < 8; ++e) {
        gr[d * 8 + e] += vr[d] * vr[e] - vi[d] * vi[e];
        gi[d * 8 + e] += vr[d] * vi[e] + vi[d] * vr[e];
      }
    }
  }
#pragma unroll
  for (int i = 0; i < 64; ++i) { float r = wave_reduce(gr[i]); if (lane == 0) s_part[wid][i] = r; }
  __syncthreads();
  if (tid < 64) { float s = 0.f; for (int w = 0; w < 8; ++w) s += s_part[w][tid]; s_gr[tid] = s; }
  __syncthreads();
#pragma unroll
  for (int i = 0; i < 64; ++i) { float r = wave_reduce(gi[i]); if (lane == 0) s_part[wid][i] = r; }
  __syncthreads();
  if (tid < 64) { float s = 0.f; for (int w = 0; w < 8; ++w) s += s_part[w][tid]; s_gi[tid] = s; }
  __syncthreads();
#pragma unroll
  for (int i = 0; i < 8; ++i) { float r = wave_reduce(nn[i]); if (lane == 0) s_part[wid][i] = r; }
  __syncthreads();
  if (tid < 8) {
    float s = 0.f; for (int w = 0; w < 8; ++w) s += s_part[w][tid];
    s_n[tid] = fmaxf(sqrtf(s), 1e-12f);
  }
  __syncthreads();
  if (tid < 8) {
    const int d = tid; const float tmp = temp[hd];
    float rr[8], ri[8], mr = -1e30f, mi = -1e30f;
#pragma unroll
    for (int e = 0; e < 8; ++e) {
      float inv = tmp / (s_n[d] * s_n[e]);
      rr[e] = s_gr[d * 8 + e] * inv;
      ri[e] = s_gi[d * 8 + e] * inv;
      mr = fmaxf(mr, rr[e]); mi = fmaxf(mi, ri[e]);
    }
    float sr = 0.f, si = 0.f;
#pragma unroll
    for (int e = 0; e < 8; ++e) {
      rr[e] = __expf(rr[e] - mr); sr += rr[e];
      ri[e] = __expf(ri[e] - mi); si += ri[e];
    }
#pragma unroll
    for (int e = 0; e < 8; ++e) { s_ar[d * 8 + e] = rr[e] / sr; s_ai[d * 8 + e] = ri[e] / si; }
  }
  __syncthreads();
  for (int n = tid; n < HWL; n += 256) {
    float vr[8], vi[8];
#pragma unroll
    for (int d = 0; d < 8; ++d) { vr[d] = rp[d][n]; vi[d] = ip[d][n]; }
#pragma unroll
    for (int d = 0; d < 8; ++d) {
      float o_r = 0.f, o_i = 0.f;
#pragma unroll
      for (int e = 0; e < 8; ++e) {
        float ar = s_ar[d * 8 + e], ai = s_ai[d * 8 + e];
        o_r += ar * vr[e] - ai * vi[e];
        o_i += ar * vi[e] + ai * vr[e];
      }
      ore[((size_t)b * CC + hd * 8 + d) * HWL + n] = o_r;
      oim[((size_t)b * CC + hd * 8 + d) * HWL + n] = o_i;
    }
  }
}

// ---------------- small elementwise kernels ----------------
// complex *= real gate (gate already sigmoid'ed by GEMM epilogue)
__global__ void k_cmul(const float* __restrict__ re, const float* __restrict__ im,
                       const float* __restrict__ g,
                       float* __restrict__ ore, float* __restrict__ oim, long n) {
  long i = (long)blockIdx.x * 256 + threadIdx.x;
  if (i >= n) return;
  float s = g[i];
  ore[i] = re[i] * s;
  oim[i] = im[i] * s;
}

// y = gelu(x)*x placed into channel-concat slot (x is a full BNxCCxHW plane)
__global__ void k_gelu_gate(const float* __restrict__ X, float* __restrict__ Y,
                            int out_ctot, int out_coff) {
  long i = (long)blockIdx.x * 256 + threadIdx.x;
  long tot = (long)BN * CC * HWL;
  if (i >= tot) return;
  int b = (int)(i / ((long)CC * HWL));
  long r = i - (long)b * CC * HWL;
  float x = X[i];
  float gl = 0.5f * x * (1.f + erff(x * 0.70710678118654752f));
  Y[((size_t)b * out_ctot + out_coff) * HWL + r] = gl * x;
}

__global__ void k_add3(const float* __restrict__ a, const float* __restrict__ b,
                       const float* __restrict__ c, float* __restrict__ o, long n) {
  long i = (long)blockIdx.x * 256 + threadIdx.x;
  if (i >= n) return;
  o[i] = a[i] + b[i] + c[i];
}

__global__ void k_add2(const float* __restrict__ a, const float* __restrict__ b,
                       float* __restrict__ o, long n) {
  long i = (long)blockIdx.x * 256 + threadIdx.x;
  if (i >= n) return;
  o[i] = a[i] + b[i];
}

// =====================================================================
// host orchestration
// =====================================================================
extern "C" void kernel_launch(void* const* d_in, const int* in_sizes, int n_in,
                              void* d_out, int out_size, void* d_ws, size_t ws_size,
                              hipStream_t stream) {
  // ---- inputs in setup_inputs() dict order ----
  const float* x          = (const float*)d_in[0];
  const float* ln0_g      = (const float*)d_in[1];
  const float* ln0_b      = (const float*)d_in[2];
  const float* ln1_g      = (const float*)d_in[3];
  const float* ln1_b      = (const float*)d_in[4];
  const float* pos_w      = (const float*)d_in[5];
  const float* pos_b      = (const float*)d_in[6];
  const float* qkv_d3_w   = (const float*)d_in[7];
  const float* qkv_d3_b   = (const float*)d_in[8];
  const float* qkv_d5_w   = (const float*)d_in[9];
  const float* qkv_d5_b   = (const float*)d_in[10];
  const float* ssa_temp   = (const float*)d_in[11];
  const float* ssa_d3_w   = (const float*)d_in[12];
  const float* ssa_d3_b   = (const float*)d_in[13];
  const float* ssa_d5_w   = (const float*)d_in[14];
  const float* ssa_d5_b   = (const float*)d_in[15];
  const float* ssa_proj_w = (const float*)d_in[16];
  const float* fsa_temp   = (const float*)d_in[17];
  const float* fsa_w0     = (const float*)d_in[18];
  const float* fsa_w1     = (const float*)d_in[19];
  const float* fsa_b1     = (const float*)d_in[20];
  const float* fsa_proj_w = (const float*)d_in[21];
  const float* f0_w0      = (const float*)d_in[22];
  const float* f0_w1      = (const float*)d_in[23];
  const float* f0_b1      = (const float*)d_in[24];
  const float* d0_w       = (const float*)d_in[25];
  const float* d0_b       = (const float*)d_in[26];
  const float* f1_w0      = (const float*)d_in[27];
  const float* f1_w1      = (const float*)d_in[28];
  const float* f1_b1      = (const float*)d_in[29];
  const float* d1_w       = (const float*)d_in[30];
  const float* d1_b       = (const float*)d_in[31];
  const float* ffn_proj_w = (const float*)d_in[32];
  (void)in_sizes; (void)n_in; (void)out_size; (void)ws_size;

  // ---- workspace layout (floats), ~25.06 planes ----
  float* ws  = (float*)d_ws;
  const size_t P = PLANE;
  float* Av   = ws;            // x0 / gate / f            (1P)
  float* B1v  = ws + 1 * P;    // pos out / effn concat lo (3P)
  float* B2v  = ws + 4 * P;    // qkv out  / effn concat hi(3P)
  float* C1v  = ws + 7 * P;    // 2C concats               (2P)
  float* Dv   = ws + 9 * P;    // ssa out / effn s / effn  (1P)
  float* Ev   = ws + 10 * P;   // fsa out                  (1P)
  float* Fv   = ws + 11 * P;   // x + ssa + fsa            (1P)
  float* Gv   = ws + 12 * P;   // ln1 out                  (1P)
  float* REv  = ws + 13 * P;   // complex temps            (2P each)
  float* IMv  = ws + 15 * P;
  float* RE2v = ws + 17 * P;
  float* IM2v = ws + 19 * P;
  float* RE3v = ws + 21 * P;
  float* IM3v = ws + 23 * P;
  float* Hv   = ws + 25 * P;   // 4-channel gate mid (P/16)

  const dim3 blk(256);
  auto gE = [](long n) { return dim3((unsigned)((n + 255) / 256)); };
  auto gGemm = [](int Cout) { return dim3(HWL / 128, (unsigned)((Cout + 15) / 16), BN); };
  auto gFft = [](int Cimg) { return dim3((unsigned)(BN * Cimg * 32)); };
  const float inv128 = 1.f / 128.f;

  // ================= LN0 =================
  k_layernorm<<<gE((long)BN * HWL), blk, 0, stream>>>(x, ln0_g, ln0_b, Av);

  // ================= SSA =================
  k_gemm1x1<<<gGemm(192), blk, 0, stream>>>(Av, pos_w, pos_b, B1v, 64, 192, 192, 0, 0);
  k_dwconv<<<gE((long)BN * 96 * HWL), blk, 0, stream>>>(B1v, qkv_d3_w, qkv_d3_b, B2v, 192, 96, 2, 3, 1, 192, 0);
  k_dwconv<<<gE((long)BN * 96 * HWL), blk, 0, stream>>>(B1v, qkv_d5_w, qkv_d5_b, B2v, 192, 96, 2, 5, 2, 192, 96);
  k_ssa_attn<<<dim3(BN * 8), blk, 0, stream>>>(B2v, ssa_temp, C1v, 128, 0);
  k_dwconv<<<gE((long)BN * 32 * HWL), blk, 0, stream>>>(Av, ssa_d3_w, ssa_d3_b, C1v, 64, 32, 2, 3, 1, 128, 64);
  k_dwconv<<<gE((long)BN * 32 * HWL), blk, 0, stream>>>(Av, ssa_d5_w, ssa_d5_b, C1v, 64, 32, 2, 5, 2, 128, 96);
  k_gemm1x1<<<gGemm(64), blk, 0, stream>>>(C1v, ssa_proj_w, nullptr, Dv, 128, 64, 64, 0, 0);

  // ================= FSA =================
  // xf = fft2(x0)
  k_fft128<<<gFft(64), blk, 0, stream>>>(Av, nullptr, RE2v, IM2v, 64, 0, +1, 1.f, 0, 64, 0);
  k_fft128<<<gFft(64), blk, 0, stream>>>(RE2v, IM2v, REv, IMv, 64, 1, +1, 1.f, 0, 64, 0);
  // a = ifft2(attn @ v), |.| into concat ch 0..63
  k_fsa_attn<<<dim3(BN * 8), blk, 0, stream>>>(REv, IMv, fsa_temp, RE2v, IM2v);
  k_fft128<<<gFft(64), blk, 0, stream>>>(RE2v, IM2v, RE3v, IM3v, 64, 0, -1, inv128, 0, 64, 0);
  k_fft128<<<gFft(64), blk, 0, stream>>>(RE3v, IM3v, C1v, nullptr, 64, 1, -1, inv128, 1, 128, 0);
  // b2 = |ifft2(sigmoid(gate(xf.re)) * xf)| into concat ch 64..127
  k_gemm1x1<<<gGemm(4), blk, 0, stream>>>(REv, fsa_w0, nullptr, Hv, 64, 4, 4, 0, 1);
  k_gemm1x1<<<gGemm(64), blk, 0, stream>>>(Hv, fsa_w1, fsa_b1, Av, 4, 64, 64, 0, 2);
  k_cmul<<<gE((long)P), blk, 0, stream>>>(REv, IMv, Av, RE2v, IM2v, (long)P);
  k_fft128<<<gFft(64), blk, 0, stream>>>(RE2v, IM2v, RE3v, IM3v, 64, 0, -1, inv128, 0, 64, 0);
  k_fft128<<<gFft(64), blk, 0, stream>>>(RE3v, IM3v, C1v, nullptr, 64, 1, -1, inv128, 1, 128, 64);
  k_gemm1x1<<<gGemm(64), blk, 0, stream>>>(C1v, fsa_proj_w, nullptr, Ev, 128, 64, 64, 0, 0);

  // ============ residual sum, LN1 ============
  k_add3<<<gE((long)P), blk, 0, stream>>>(x, Dv, Ev, Fv, (long)P);
  k_layernorm<<<gE((long)BN * HWL), blk, 0, stream>>>(Fv, ln1_g, ln1_b, Gv);

  // ================= EFFN =================
  // f = freq_percep(x1)  -> Av
  k_fft128<<<gFft(64), blk, 0, stream>>>(Gv, nullptr, RE2v, IM2v, 64, 0, +1, 1.f, 0, 64, 0);
  k_fft128<<<gFft(64), blk, 0, stream>>>(RE2v, IM2v, REv, IMv, 64, 1, +1, 1.f, 0, 64, 0);
  k_gemm1x1<<<gGemm(4), blk, 0, stream>>>(REv, f0_w0, nullptr, Hv, 64, 4, 4, 0, 1);
  k_gemm1x1<<<gGemm(64), blk, 0, stream>>>(Hv, f0_w1, f0_b1, Av, 4, 64, 64, 0, 2);
  k_cmul<<<gE((long)P), blk, 0, stream>>>(REv, IMv, Av, RE2v, IM2v, (long)P);
  k_fft128<<<gFft(64), blk, 0, stream>>>(RE2v, IM2v, RE3v, IM3v, 64, 0, -1, inv128, 0, 64, 0);
  k_fft128<<<gFft(64), blk, 0, stream>>>(RE3v, IM3v, Av, nullptr, 64, 1, -1, inv128, 1, 64, 0);
  // s = dwconv3x3(x1) -> Dv
  k_dwconv<<<gE((long)BN * 64 * HWL), blk, 0, stream>>>(Gv, d0_w, d0_b, Dv, 64, 64, 1, 3, 1, 64, 0);
  // y = [gelu(f)*f | gelu(s)*s] -> C1 (128 ch)
  k_gelu_gate<<<gE((long)P), blk, 0, stream>>>(Av, C1v, 128, 0);
  k_gelu_gate<<<gE((long)P), blk, 0, stream>>>(Dv, C1v, 128, 64);
  // freq_percep(y) (128 ch) -> effn concat ch 0..127 at B1
  k_fft128<<<gFft(128), blk, 0, stream>>>(C1v, nullptr, RE2v, IM2v, 128, 0, +1, 1.f, 0, 128, 0);
  k_fft128<<<gFft(128), blk, 0, stream>>>(RE2v, IM2v, REv, IMv, 128, 1, +1, 1.f, 0, 128, 0);
  k_gemm1x1<<<gGemm(4), blk, 0, stream>>>(REv, f1_w0, nullptr, Hv, 128, 4, 4, 0, 1);
  k_gemm1x1<<<gGemm(128), blk, 0, stream>>>(Hv, f1_w1, f1_b1, RE3v, 4, 128, 128, 0, 2);
  k_cmul<<<gE((long)(2 * P)), blk, 0, stream>>>(REv, IMv, RE3v, RE2v, IM2v, (long)(2 * P));
  k_fft128<<<gFft(128), blk, 0, stream>>>(RE2v, IM2v, RE3v, IM3v, 128, 0, -1, inv128, 0, 128, 0);
  k_fft128<<<gFft(128), blk, 0, stream>>>(RE3v, IM3v, B1v, nullptr, 128, 1, -1, inv128, 1, 256, 0);
  // dwconv(y) -> effn concat ch 128..255
  k_dwconv<<<gE((long)BN * 128 * HWL), blk, 0, stream>>>(C1v, d1_w, d1_b, B1v, 128, 128, 2, 3, 1, 256, 128);
  // ffn projection 256 -> 64
  k_gemm1x1<<<gGemm(64), blk, 0, stream>>>(B1v, ffn_proj_w, nullptr, Dv, 256, 64, 64, 0, 0);

  // ============ final residual ============
  k_add2<<<gE((long)P), blk, 0, stream>>>(Fv, Dv, (float*)d_out, (long)P);
}
